// RNN_7499012899285
// MI455X (gfx1250) — matrix operand order repaired and checked by reading, not compiled, not run on hardware
//
#include <hip/hip_runtime.h>
#include <stdint.h>

#define TT 16384
#define HH 1120
#define H3 3360
#define NM 32

typedef __attribute__((ext_vector_type(16))) __bf16 v16bf;
typedef __attribute__((ext_vector_type(8)))  float  v8f;

// ---------- helpers ----------
__device__ __forceinline__ uint16_t f2bf(float f) {
  union { float f; uint32_t u; } c; c.f = f;
  uint32_t u = c.u;
  uint32_t r = (u + 0x7FFFu + ((u >> 16) & 1u)) >> 16;  // RNE fp32->bf16
  return (uint16_t)r;
}

__global__ void cvt_f32_bf16(const float* __restrict__ in, uint16_t* __restrict__ out, long n) {
  long i = (long)blockIdx.x * blockDim.x + threadIdx.x;
  long stride = (long)gridDim.x * blockDim.x;
  for (; i < n; i += stride) out[i] = f2bf(in[i]);
}

// async 16B global->LDS transfer (ASYNCcnt-tracked, no VGPR round-trip)
__device__ __forceinline__ void async_copy16(const void* gsrc, void* lds_dst) {
  uint32_t ldst = (uint32_t)(size_t)lds_dst;  // low 32 bits = LDS byte offset
  asm volatile("global_load_async_to_lds_b128 %0, %1, off"
               :: "v"(ldst), "v"(gsrc) : "memory");
}

// ---------- bf16 WMMA GEMM:  C[M,N] = A[M,K] * B[N,K]^T + bias[N] ----------
// Block: 256 threads (8 waves). Block tile: 512(M) x 32(N). Wave tile: 64x32 (4x2 WMMA).
// Both tiles are staged with global_load_async_to_lds_b128. The A tile is stored with
// 8-column groups swizzled (0,2,1,3) so each lane's 16-element WMMA fragment is
// contiguous -> two ds_load_b128 per fragment for both A and B.
__global__ void __launch_bounds__(256) wmma_gemm_bf16(
    const uint16_t* __restrict__ A,   // [M,K] bf16 row-major (activations)
    const uint16_t* __restrict__ B,   // [N,K] bf16 row-major (weights W; C = A @ W^T)
    const float*    __restrict__ bias,// [N]
    float*          __restrict__ C,   // [M,N] fp32
    int M, int N, int K)
{
  __shared__ __align__(16) uint16_t Asb[512 * 32];  // 32 KB
  __shared__ __align__(16) uint16_t Bsb[32 * 32];   //  2 KB

  const int tid  = threadIdx.x;
  const int lane = tid & 31;
  const int wave = tid >> 5;
  const int half = lane >> 4;
  const int l15  = lane & 15;
  const int mBase = blockIdx.x * 512;
  const int nBase = blockIdx.y * 32;
  const int mW = wave * 64;

  v8f acc[4][2];
#pragma unroll
  for (int i = 0; i < 4; ++i)
#pragma unroll
    for (int j = 0; j < 2; ++j) acc[i][j] = (v8f){};

  union Frag { v16bf v; uint4 q[2]; };

  for (int kc = 0; kc < K; kc += 32) {
    // ---- stage A tile (512 x 32 bf16): 8 async 16B chunks/thread, swizzle s->{0,2,1,3}
#pragma unroll
    for (int it = 0; it < 8; ++it) {
      int li = tid + 256 * it;
      int r = li >> 2, s = li & 3;
      int sp = ((s & 1) << 1) | (s >> 1);
      async_copy16(A + (size_t)(mBase + r) * K + kc + s * 8, &Asb[r * 32 + sp * 8]);
    }
    // ---- stage B tile (32 x 32 bf16): 1 async chunk for half the threads
    if (tid < 128) {
      int r = tid >> 2, s = tid & 3;
      async_copy16(B + (size_t)(nBase + r) * K + kc + s * 8, &Bsb[r * 32 + (((s & 1) << 1) | (s >> 1)) * 8]);
    }
    if (kc + 32 < K)  // pull next A chunk toward L2 (global_prefetch_b8)
      __builtin_prefetch(A + (size_t)(mBase + tid * 2) * K + kc + 32, 0, 1);
    asm volatile("s_wait_asynccnt 0x0" ::: "memory");
    __syncthreads();

    // ---- fragments: contiguous 32B per lane -> two ds_load_b128 each
    Frag bf0, bf1;
    bf0.q[0] = *(const uint4*)&Bsb[(l15)      * 32 + half * 16];
    bf0.q[1] = *(const uint4*)&Bsb[(l15)      * 32 + half * 16 + 8];
    bf1.q[0] = *(const uint4*)&Bsb[(16 + l15) * 32 + half * 16];
    bf1.q[1] = *(const uint4*)&Bsb[(16 + l15) * 32 + half * 16 + 8];

#pragma unroll
    for (int mi = 0; mi < 4; ++mi) {
      Frag af;
      int row = mW + mi * 16 + l15;
      af.q[0] = *(const uint4*)&Asb[row * 32 + half * 16];
      af.q[1] = *(const uint4*)&Asb[row * 32 + half * 16 + 8];
      acc[mi][0] = __builtin_amdgcn_wmma_f32_16x16x32_bf16(false, af.v, false, bf0.v,
                                                           (short)0, acc[mi][0], false, false);
      acc[mi][1] = __builtin_amdgcn_wmma_f32_16x16x32_bf16(false, af.v, false, bf1.v,
                                                           (short)0, acc[mi][1], false, false);
    }
    __syncthreads();
  }

  // ---- epilogue: C layout -> lane l15 = N, VGPR v -> M = v + 8*half
  int n0 = nBase + l15;
  int n1 = nBase + 16 + l15;
  float bv0 = bias[n0], bv1 = bias[n1];
#pragma unroll
  for (int mi = 0; mi < 4; ++mi) {
#pragma unroll
    for (int v = 0; v < 8; ++v) {
      int m = mBase + mW + mi * 16 + v + half * 8;
      C[(size_t)m * N + n0] = acc[mi][0][v] + bv0;
      C[(size_t)m * N + n1] = acc[mi][1][v] + bv1;
    }
  }
}

// ---------- grid-wide split barrier (generation counter, agent scope) ----------
__device__ __forceinline__ void grid_barrier(unsigned* cnt, unsigned* gen, unsigned nwg) {
  __syncthreads();
  if (threadIdx.x == 0) {
    unsigned g = __hip_atomic_load(gen, __ATOMIC_RELAXED, __HIP_MEMORY_SCOPE_AGENT);
    __threadfence();
    unsigned arrived = __hip_atomic_fetch_add(cnt, 1u, __ATOMIC_ACQ_REL, __HIP_MEMORY_SCOPE_AGENT);
    if (arrived == nwg - 1u) {
      __hip_atomic_store(cnt, 0u, __ATOMIC_RELAXED, __HIP_MEMORY_SCOPE_AGENT);
      __hip_atomic_store(gen, g + 1u, __ATOMIC_RELEASE, __HIP_MEMORY_SCOPE_AGENT);
    } else {
      while (__hip_atomic_load(gen, __ATOMIC_ACQUIRE, __HIP_MEMORY_SCOPE_AGENT) == g)
        __builtin_amdgcn_s_sleep(2);
    }
  }
  __syncthreads();
}

// ---------- persistent GRU scan over T steps ----------
// Grid: HH/4 = 280 WGs x 256 threads. WG owns outputs j = jb..jb+3 and caches the
// 12 matching rows of W_hh (r, H+r, 2H+r) in LDS.
// Rows are stored as 32 lane-chunks of 35 floats padded to 36 (144 B, 16B-aligned)
// so the inner dot product vectorizes to ds_load_b128 (stride 36 banks: conflict-free).
__global__ void __launch_bounds__(256) gru_scan(
    const float* __restrict__ Whh,  // [3H, H] fp32
    const float* __restrict__ ig,   // [T, 3H] fp32 precomputed input gates (incl. b)
    const float* __restrict__ bn,   // [H]
    float*       hvec,              // [H] fp32 recurrent state (workspace)
    uint16_t*    hout,              // [T, H] bf16 layer output
    unsigned* syncCnt, unsigned* syncGen)
{
  __shared__ __align__(16) float wlds[12 * 1152];  // 55.3 KB (row stride 32*36)
  __shared__ __align__(16) float hpad[32 * 36];    //  4.6 KB
  __shared__ float hg[12];

  const int tid  = threadIdx.x;
  const int lane = tid & 31;
  const int wv   = tid >> 5;
  const int jb   = blockIdx.x * 4;
  const unsigned nwg = gridDim.x;

  // one-time: load this WG's 12 rows of W_hh into LDS (padded per-lane chunks)
  for (int idx = tid; idx < 12 * HH; idx += 256) {
    int lr = idx / HH, rem = idx - lr * HH;
    int ln = rem / 35, o = rem - ln * 35;
    int sec = lr >> 2, tj = lr & 3;
    wlds[lr * 1152 + ln * 36 + o] = Whh[(size_t)(sec * HH + jb + tj) * HH + rem];
  }
  if (tid < 4)
    __hip_atomic_store(&hvec[jb + tid], 0.f, __ATOMIC_RELEASE, __HIP_MEMORY_SCOPE_AGENT);
  grid_barrier(syncCnt, syncGen, nwg);

  for (int t = 0; t < TT; ++t) {
    // stage full h into padded LDS, then slice contiguous 35-float chunk into registers
    for (int i = tid; i < HH; i += 256) {
      int ln = i / 35, o = i - ln * 35;
      hpad[ln * 36 + o] = hvec[i];
    }
    __syncthreads();
    float hreg[35];
#pragma unroll
    for (int i = 0; i < 35; ++i) hreg[i] = hpad[lane * 36 + i];

    // 12 dot products of length 1120; wave-per-row, lane-sliced, shfl reduce
    for (int lr = wv; lr < 12; lr += 8) {
      const float* wr = &wlds[lr * 1152 + lane * 36];
      float s = 0.f;
#pragma unroll
      for (int i = 0; i < 35; ++i) s = fmaf(wr[i], hreg[i], s);
#pragma unroll
      for (int off = 16; off > 0; off >>= 1) s += __shfl_xor(s, off, 32);
      if (lane == 0) hg[lr] = s;
    }
    __syncthreads();

    // gate math for our 4 outputs
    if (tid < 4) {
      int j = jb + tid;
      const float* igt = ig + (size_t)t * H3;
      float rg = 1.f / (1.f + __expf(-(igt[j]      + hg[tid])));
      float zg = 1.f / (1.f + __expf(-(igt[HH + j] + hg[4 + tid])));
      float ng = tanhf(igt[2 * HH + j] + rg * (hg[8 + tid] + bn[j]));
      int ln = j / 35, o = j - ln * 35;
      float hp = hpad[ln * 36 + o];
      float hn = ng + zg * (hp - ng);
      __hip_atomic_store(&hvec[j], hn, __ATOMIC_RELEASE, __HIP_MEMORY_SCOPE_AGENT);
      hout[(size_t)t * HH + j] = f2bf(hn);
    }
    grid_barrier(syncCnt, syncGen, nwg);
  }
}

// ---------- host ----------
extern "C" void kernel_launch(void* const* d_in, const int* in_sizes, int n_in,
                              void* d_out, int out_size, void* d_ws, size_t ws_size,
                              hipStream_t stream) {
  (void)in_sizes; (void)n_in; (void)out_size; (void)ws_size;

  const float* y    = (const float*)d_in[0];
  const float* Wih0 = (const float*)d_in[1];
  const float* Whh0 = (const float*)d_in[2];
  const float* b0   = (const float*)d_in[3];
  const float* bn0  = (const float*)d_in[4];
  const float* Wih1 = (const float*)d_in[5];
  const float* Whh1 = (const float*)d_in[6];
  const float* b1   = (const float*)d_in[7];
  const float* bn1  = (const float*)d_in[8];
  const float* Wlin = (const float*)d_in[9];
  const float* blin = (const float*)d_in[10];

  char* ws = (char*)d_ws;
  size_t off = 0;
  auto alloc = [&](size_t bytes) { size_t r = off; off += (bytes + 255) & ~(size_t)255; return r; };
  size_t ig_off   = alloc((size_t)TT * H3 * 4);   // shared by both layers
  size_t ybf_off  = alloc((size_t)TT * NM * 2);
  size_t h1_off   = alloc((size_t)TT * HH * 2);
  size_t h2_off   = alloc((size_t)TT * HH * 2);
  size_t wih0_off = alloc((size_t)H3 * NM * 2);
  size_t wih1_off = alloc((size_t)H3 * HH * 2);
  size_t wlin_off = alloc((size_t)HH * HH * 2);
  size_t hvec_off = alloc((size_t)HH * 4);
  size_t sync_off = alloc(256);

  float*    ig     = (float*)(ws + ig_off);
  uint16_t* ybf    = (uint16_t*)(ws + ybf_off);
  uint16_t* h1bf   = (uint16_t*)(ws + h1_off);
  uint16_t* h2bf   = (uint16_t*)(ws + h2_off);
  uint16_t* wih0bf = (uint16_t*)(ws + wih0_off);
  uint16_t* wih1bf = (uint16_t*)(ws + wih1_off);
  uint16_t* wlinbf = (uint16_t*)(ws + wlin_off);
  float*    hvec   = (float*)(ws + hvec_off);
  unsigned* syncC  = (unsigned*)(ws + sync_off);
  unsigned* syncG  = syncC + 16;

  // fp32 -> bf16 packs
  cvt_f32_bf16<<<256, 256, 0, stream>>>(y,    ybf,    (long)TT * NM);
  cvt_f32_bf16<<<64,  256, 0, stream>>>(Wih0, wih0bf, (long)H3 * NM);
  cvt_f32_bf16<<<1024,256, 0, stream>>>(Wih1, wih1bf, (long)H3 * HH);
  cvt_f32_bf16<<<512, 256, 0, stream>>>(Wlin, wlinbf, (long)HH * HH);

  // igates0 = y @ W_ih0^T + b0   (M=16384, N=3360, K=32)
  wmma_gemm_bf16<<<dim3(TT / 512, H3 / 32), 256, 0, stream>>>(ybf, wih0bf, b0, ig, TT, H3, NM);

  // layer 0 recurrence (persistent, grid-synced)
  hipMemsetAsync(ws + sync_off, 0, 256, stream);
  gru_scan<<<HH / 4, 256, 0, stream>>>(Whh0, ig, bn0, hvec, h1bf, syncC, syncG);

  // igates1 = h1 @ W_ih1^T + b1  (K=1120)
  wmma_gemm_bf16<<<dim3(TT / 512, H3 / 32), 256, 0, stream>>>(h1bf, wih1bf, b1, ig, TT, H3, HH);

  // layer 1 recurrence
  hipMemsetAsync(ws + sync_off, 0, 256, stream);
  gru_scan<<<HH / 4, 256, 0, stream>>>(Whh1, ig, bn1, hvec, h2bf, syncC, syncG);

  // out = h2 @ W_lin^T + b_lin   (N=1120, K=1120)
  wmma_gemm_bf16<<<dim3(TT / 512, HH / 32), 256, 0, stream>>>(h2bf, wlinbf, blin, (float*)d_out, TT, HH, HH);
}
